// SlotAttention_82471962018311
// MI455X (gfx1250) — matrix operand order repaired and checked by reading, not compiled, not run on hardware
//
#include <hip/hip_runtime.h>

// ---------------- types / helpers ----------------
typedef __attribute__((ext_vector_type(16))) __bf16 v16bf;
typedef __attribute__((ext_vector_type(8)))  float  v8f;

static __device__ inline unsigned short f2bf(float x) {
    union { float f; unsigned u; } a; a.f = x;
    unsigned r = a.u + 0x7FFFu + ((a.u >> 16) & 1u);   // round-to-nearest-even
    return (unsigned short)(r >> 16);
}
static __device__ inline unsigned pack2(unsigned short lo, unsigned short hi) {
    return (unsigned)lo | ((unsigned)hi << 16);
}
static __device__ inline v8f wmma_bf16(v16bf a, v16bf b, v8f c) {
    return __builtin_amdgcn_wmma_f32_16x16x32_bf16(false, a, false, b, (short)0, c, false, false);
}
// CDNA5 async copy global->LDS (ASYNCcnt), 16B per active lane
static __device__ inline void async_load_b128(unsigned lds_off, const void* gptr) {
    asm volatile("global_load_async_to_lds_b128 %0, %1, off"
                 :: "v"(lds_off), "v"((unsigned long long)(size_t)gptr)
                 : "memory");
}
static __device__ inline void wait_async() {
    asm volatile("s_wait_asynccnt 0x0" ::: "memory");
}
// low 32 bits of a generic pointer into LDS = LDS byte offset (flat-aperture rule)
static __device__ inline unsigned lds_off32(const void* p) {
    return (unsigned)(unsigned long long)(size_t)p;
}
// A fragment: 16x32 bf16 tile, row-major ushort source (row = lane&15)
static __device__ inline v16bf load_frag_a(const unsigned short* p, int stride, int lane) {
    int hf = (lane >> 4) & 1, r = lane & 15;
    union { v16bf v; unsigned u[8]; } f;
    const unsigned short* row = p + r * stride;
#pragma unroll
    for (int i = 0; i < 4; i++) {
        int k0 = hf * 16 + 2 * i;
        f.u[i]     = *(const unsigned*)(row + k0);
        f.u[i + 4] = *(const unsigned*)(row + 8 + k0);
    }
    return f.v;
}
// B fragment: 32x16 bf16 tile stored TRANSPOSED [col][k] (col = lane&15)
static __device__ inline v16bf load_frag_b(const unsigned short* p, int stride, int lane) {
    int hf = (lane >> 4) & 1, r = lane & 15;
    union { v16bf v; unsigned u[8]; } f;
    const unsigned short* col = p + r * stride + hf * 16;
#pragma unroll
    for (int i = 0; i < 8; i++) f.u[i] = *(const unsigned*)(col + 2 * i);
    return f.v;
}

#define BB 64
#define CC 512
#define NN 1024
#define DD 128
#define KK 8

// ============ Kernel 0: one-time weight prep: bf16 transposed GEMM weights + f32 transposed GRU weights ============
__global__ __launch_bounds__(256) void k_prep_w(
    const float* __restrict__ Wp, const float* __restrict__ Wk,
    const float* __restrict__ Wv, const float* __restrict__ Wq,
    const float* __restrict__ W_ih, const float* __restrict__ W_hh,
    unsigned short* __restrict__ WpT, unsigned short* __restrict__ WkT,
    unsigned short* __restrict__ WvT, unsigned short* __restrict__ WqT,
    float* __restrict__ W_ihT, float* __restrict__ W_hhT)
{
    int i = blockIdx.x * 256 + threadIdx.x;
    if (i < CC * DD) {                       // WpT[d][c] = Wp[c][d]
        int d = i >> 9, c = i & 511;
        WpT[i] = f2bf(Wp[(size_t)c * DD + d]);
    }
    if (i < DD * DD) {                       // 128x128 transposes
        int d = i >> 7, c = i & 127;
        WkT[i] = f2bf(Wk[(size_t)c * DD + d]);
        WvT[i] = f2bf(Wv[(size_t)c * DD + d]);
        WqT[i] = f2bf(Wq[(size_t)c * DD + d]);
    }
    if (i < DD * 3 * DD) {                   // W_ihT[d][g] = W_ih[g][d]
        int d = i / 384, g = i % 384;
        W_ihT[i] = W_ih[(size_t)g * DD + d];
        W_hhT[i] = W_hh[(size_t)g * DD + d];
    }
}

// ============ Kernel 1: inputs = LN(x_flat@Wp+bp); k = in@Wk+bk; v = in@Wv+bv ============
__global__ __launch_bounds__(256) void k_input_proj(
    const float* __restrict__ x, const unsigned short* __restrict__ WpT, const float* __restrict__ bp,
    const float* __restrict__ g_in, const float* __restrict__ b_in,
    const unsigned short* __restrict__ WkT, const float* __restrict__ bk,
    const unsigned short* __restrict__ WvT, const float* __restrict__ bv,
    unsigned short* __restrict__ k_bf, unsigned short* __restrict__ v_t)
{
    __shared__ unsigned short lA[16 * 36];     // x tile  [m][c]
    __shared__ unsigned short lB[128 * 40];    // weight tile [dout][k], row stride 80B (16B aligned)
    __shared__ float          lO[16 * 132];    // fp32 proj result [m][d]
    __shared__ unsigned short lI[16 * 132];    // bf16 LN'd inputs [m][d]
    __shared__ float mu[16], rs[16];

    int t = threadIdx.x, wave = t >> 5, lane = t & 31;
    int hf = lane >> 4, r = lane & 15;
    int m0 = blockIdx.x * 16;
    int b = m0 / NN, n0 = m0 % NN;
    int d0 = wave * 16;
    unsigned lb0 = lds_off32(lB);

    // ---- GEMM1: [16 x 512] x [512 x 128] over K-chunks of 32 ----
    v8f acc = {};
    for (int c0 = 0; c0 < CC; c0 += 32) {
        // async weight tile: WpT rows, 32 bf16 per row = 4 x b128 segments
        for (int sidx = t; sidx < 512; sidx += 256) {
            int d = sidx >> 2, s = sidx & 3;
            async_load_b128(lb0 + (unsigned)(d * 80 + s * 16),
                            WpT + (size_t)d * CC + c0 + s * 8);
        }
        // x tile via b128 loads: 32 c-rows x 16 n, contiguous in n
        if (t < 128) {
            int c = t >> 2, sg = t & 3;
            const float4 xv = *(const float4*)(x + (size_t)b * CC * NN + (size_t)(c0 + c) * NN + n0 + sg * 4);
            lA[(sg * 4 + 0) * 36 + c] = f2bf(xv.x);
            lA[(sg * 4 + 1) * 36 + c] = f2bf(xv.y);
            lA[(sg * 4 + 2) * 36 + c] = f2bf(xv.z);
            lA[(sg * 4 + 3) * 36 + c] = f2bf(xv.w);
        }
        wait_async();
        __syncthreads();
        v16bf af = load_frag_a(lA, 36, lane);
        v16bf bf = load_frag_b(lB + d0 * 40, 40, lane);
        acc = wmma_bf16(af, bf, acc);
        __syncthreads();
    }
#pragma unroll
    for (int i = 0; i < 8; i++) lO[(i + 8 * hf) * 132 + d0 + r] = acc[i] + bp[d0 + r];
    __syncthreads();

    // ---- LayerNorm over D per row ----
    if (t < 16) {
        float s = 0.f, s2 = 0.f;
        for (int d = 0; d < DD; d++) { float v = lO[t * 132 + d]; s += v; s2 += v * v; }
        float m = s / DD;
        mu[t] = m; rs[t] = rsqrtf(s2 / DD - m * m + 1e-5f);
    }
    __syncthreads();
    for (int idx = t; idx < 16 * DD; idx += 256) {
        int row = idx >> 7, d = idx & 127;
        lI[row * 132 + d] = f2bf((lO[row * 132 + d] - mu[row]) * rs[row] * g_in[d] + b_in[d]);
    }
    __syncthreads();

    // ---- k and v projections: [16 x 128] x [128 x 128] ----
    for (int which = 0; which < 2; which++) {
        const unsigned short* Wt = which ? WvT : WkT;
        const float* bt = which ? bv : bk;
        v8f a2 = {};
        for (int c0 = 0; c0 < DD; c0 += 32) {
            for (int sidx = t; sidx < 512; sidx += 256) {
                int d = sidx >> 2, s = sidx & 3;
                async_load_b128(lb0 + (unsigned)(d * 80 + s * 16),
                                Wt + (size_t)d * DD + c0 + s * 8);
            }
            wait_async();
            __syncthreads();
            v16bf af = load_frag_a(lI + c0, 132, lane);
            v16bf bf = load_frag_b(lB + d0 * 40, 40, lane);
            a2 = wmma_bf16(af, bf, a2);
            __syncthreads();
        }
        if (which == 0) {      // k: [b][n][d]
#pragma unroll
            for (int i = 0; i < 8; i++)
                k_bf[(size_t)(m0 + i + 8 * hf) * DD + d0 + r] = f2bf(a2[i] + bt[d0 + r]);
        } else {               // v: transposed [b][d][n] for fast B-fragments in attention
#pragma unroll
            for (int i = 0; i < 8; i++)
                v_t[((size_t)b * DD + d0 + r) * NN + n0 + i + 8 * hf] = f2bf(a2[i] + bt[d0 + r]);
        }
        __syncthreads();
    }
}

// ============ Kernel 2: slots = mu + exp(log_sigma)*noise ============
__global__ void k_slots_init(const float* __restrict__ smu, const float* __restrict__ sls,
                             const float* __restrict__ noise, float* __restrict__ slots)
{
    int i = blockIdx.x * blockDim.x + threadIdx.x;
    if (i < BB * KK * DD) {
        int kd = i % (KK * DD);
        slots[i] = smu[kd] + expf(sls[kd]) * noise[i];
    }
}

// ============ Kernel 3: q = LN(slots)@Wq + bq  (bf16 out) ============
__global__ __launch_bounds__(256) void k_qproj(
    const float* __restrict__ slots, const float* __restrict__ gsl, const float* __restrict__ bsl,
    const unsigned short* __restrict__ WqT, const float* __restrict__ bq,
    unsigned short* __restrict__ q_bf)
{
    __shared__ float          lO[16 * 132];
    __shared__ unsigned short lI[16 * 132];
    __shared__ unsigned short lB[128 * 40];
    __shared__ float mu[16], rs[16];
    int t = threadIdx.x, wave = t >> 5, lane = t & 31;
    int hf = lane >> 4, r = lane & 15;
    int d0 = wave * 16, m0 = blockIdx.x * 16;
    unsigned lb0 = lds_off32(lB);

    for (int idx = t; idx < 16 * DD; idx += 256) {
        int row = idx >> 7, d = idx & 127;
        lO[row * 132 + d] = slots[(size_t)(m0 + row) * DD + d];
    }
    __syncthreads();
    if (t < 16) {
        float s = 0.f, s2 = 0.f;
        for (int d = 0; d < DD; d++) { float v = lO[t * 132 + d]; s += v; s2 += v * v; }
        float m = s / DD;
        mu[t] = m; rs[t] = rsqrtf(s2 / DD - m * m + 1e-5f);
    }
    __syncthreads();
    for (int idx = t; idx < 16 * DD; idx += 256) {
        int row = idx >> 7, d = idx & 127;
        lI[row * 132 + d] = f2bf((lO[row * 132 + d] - mu[row]) * rs[row] * gsl[d] + bsl[d]);
    }
    __syncthreads();
    v8f acc = {};
    for (int c0 = 0; c0 < DD; c0 += 32) {
        for (int sidx = t; sidx < 512; sidx += 256) {
            int d = sidx >> 2, s = sidx & 3;
            async_load_b128(lb0 + (unsigned)(d * 80 + s * 16),
                            WqT + (size_t)d * DD + c0 + s * 8);
        }
        wait_async();
        __syncthreads();
        v16bf af = load_frag_a(lI + c0, 132, lane);
        v16bf bf = load_frag_b(lB + d0 * 40, 40, lane);
        acc = wmma_bf16(af, bf, acc);
        __syncthreads();
    }
#pragma unroll
    for (int i = 0; i < 8; i++)
        q_bf[(size_t)(m0 + i + 8 * hf) * DD + d0 + r] = f2bf(acc[i] + bq[d0 + r]);
}

// ============ Kernel 4: logits -> softmax(slots) -> attn_norm @ v  (one WG per batch) ============
__global__ __launch_bounds__(256) void k_attn(
    const unsigned short* __restrict__ q_bf, const unsigned short* __restrict__ k_bf,
    const unsigned short* __restrict__ v_t, float* __restrict__ attn_out,
    float* __restrict__ updates)
{
    __shared__ float sAttn[KK * NN];        // 32 KB: attn for phase B
    __shared__ float partial[8][KK];        // deterministic per-wave row sums
    __shared__ float rowsum[KK];
    int t = threadIdx.x, wave = t >> 5, lane = t & 31;
    int hf = lane >> 4, r = lane & 15, rr = r & 7;
    int b = blockIdx.x;
    const float scale = 0.088388347648318447f;  // 128^-0.5

    // hoisted A fragments (q rows, 4 K-chunks of 32 over D=128)
    union { v16bf v; unsigned u[8]; } afr[4];
    const unsigned short* qrow = q_bf + ((size_t)b * KK + rr) * DD;
#pragma unroll
    for (int c = 0; c < 4; c++)
#pragma unroll
        for (int i = 0; i < 4; i++) {
            int k0 = c * 32 + hf * 16 + 2 * i;
            afr[c].u[i]     = *(const unsigned*)(qrow + k0);
            afr[c].u[i + 4] = *(const unsigned*)(qrow + 8 + k0);
        }

    float wsum[KK];
#pragma unroll
    for (int i = 0; i < KK; i++) wsum[i] = 0.f;

    // ---- phase A: logits tiles + softmax over slots (K=8) ----
    for (int tile = wave; tile < NN / 16; tile += 8) {
        int n0 = tile * 16;
        v8f acc = {};
#pragma unroll
        for (int c = 0; c < 4; c++) {
            union { v16bf v; unsigned u[8]; } bf;
            const unsigned short* krow = k_bf + ((size_t)b * NN + n0 + r) * DD + c * 32 + hf * 16;
#pragma unroll
            for (int i = 0; i < 8; i++) bf.u[i] = *(const unsigned*)(krow + 2 * i);
            acc = wmma_bf16(afr[c].v, bf.v, acc);
        }
        if (hf == 0) {  // lanes 0-15: column n = n0+r, rows 0..7 in acc[0..7]
            float e[KK], mx = -1e30f;
#pragma unroll
            for (int i = 0; i < KK; i++) { e[i] = acc[i] * scale; mx = fmaxf(mx, e[i]); }
            float s = 0.f;
#pragma unroll
            for (int i = 0; i < KK; i++) { e[i] = expf(e[i] - mx); s += e[i]; }
            float inv = 1.f / s;
#pragma unroll
            for (int i = 0; i < KK; i++) {
                float a = e[i] * inv;
                sAttn[i * NN + n0 + r] = a;
                attn_out[((size_t)b * KK + i) * NN + n0 + r] = a;
                float v = a;
                for (int off = 8; off >= 1; off >>= 1) v += __shfl_xor(v, off, 16);
                wsum[i] += v;   // identical across lanes 0-15
            }
        }
    }
    if (hf == 0 && r == 0)
#pragma unroll
        for (int i = 0; i < KK; i++) partial[wave][i] = wsum[i];
    __syncthreads();
    if (t < KK) {
        float s = 0.f;
        for (int w = 0; w < 8; w++) s += partial[w][t];   // fixed order -> deterministic
        rowsum[t] = s;
    }
    __syncthreads();

    // ---- phase B: updates = attn_norm @ v, each wave owns a 16-wide d tile ----
    int d0 = wave * 16;
    float rnorm = 1.0f / (rowsum[rr] + 1e-8f);
    const float* arow = sAttn + rr * NN;
    const unsigned short* vrow = v_t + ((size_t)b * DD + d0 + r) * NN;
    v8f acc = {};
    for (int n0 = 0; n0 < NN; n0 += 32) {
        union { v16bf v; unsigned u[8]; } af;
#pragma unroll
        for (int i = 0; i < 4; i++) {
            int kA = n0 + hf * 16 + 2 * i;
            int kB = n0 + 8 + hf * 16 + 2 * i;
            af.u[i]     = pack2(f2bf(arow[kA] * rnorm), f2bf(arow[kA + 1] * rnorm));
            af.u[i + 4] = pack2(f2bf(arow[kB] * rnorm), f2bf(arow[kB + 1] * rnorm));
        }
        union { v16bf v; unsigned u[8]; } bf;
#pragma unroll
        for (int i = 0; i < 8; i++)
            bf.u[i] = *(const unsigned*)(vrow + n0 + hf * 16 + 2 * i);
        acc = wmma_bf16(af.v, bf.v, acc);
    }
    if (hf == 0)
#pragma unroll
        for (int i = 0; i < KK; i++)
            updates[((size_t)b * KK + i) * DD + d0 + r] = acc[i];
}

// ============ Kernel 5: GRUCell + LN + MLP + residual (one WG per slot row) ============
__global__ __launch_bounds__(128) void k_gru_mlp(
    const float* __restrict__ upd, const float* __restrict__ hprev,
    const float* __restrict__ W_ihT, const float* __restrict__ W_hhT,
    const float* __restrict__ b_ih, const float* __restrict__ b_hh,
    const float* __restrict__ W1, const float* __restrict__ b1,
    const float* __restrict__ W2, const float* __restrict__ b2,
    const float* __restrict__ g_ml, const float* __restrict__ b_ml,
    float* __restrict__ out)
{
    __shared__ float su[128], sh[128], sS[128], sLN[128], sM[256];
    __shared__ float ws1[4], ws2[4];
    int m = blockIdx.x, t = threadIdx.x;
    su[t] = upd[(size_t)m * DD + t];
    sh[t] = hprev[(size_t)m * DD + t];
    __syncthreads();

    float xr = b_ih[t], xz = b_ih[128 + t], xn = b_ih[256 + t];
    float hr = b_hh[t], hz = b_hh[128 + t], hn = b_hh[256 + t];
    for (int d = 0; d < 128; d++) {          // coalesced: W*T[d][gate*128 + t]
        float u = su[d], h = sh[d];
        const float* wi = W_ihT + (size_t)d * 384;
        const float* wh = W_hhT + (size_t)d * 384;
        xr += u * wi[t]; xz += u * wi[128 + t]; xn += u * wi[256 + t];
        hr += h * wh[t]; hz += h * wh[128 + t]; hn += h * wh[256 + t];
    }
    float rg = 1.f / (1.f + expf(-(xr + hr)));
    float zg = 1.f / (1.f + expf(-(xz + hz)));
    float ng = tanhf(xn + rg * hn);
    float s = (1.f - zg) * ng + zg * sh[t];
    sS[t] = s;
    __syncthreads();

    // LayerNorm over 128
    float v = s, v2 = s * s;
    for (int off = 16; off >= 1; off >>= 1) { v += __shfl_xor(v, off, 32); v2 += __shfl_xor(v2, off, 32); }
    if ((t & 31) == 0) { ws1[t >> 5] = v; ws2[t >> 5] = v2; }
    __syncthreads();
    float sm = ws1[0] + ws1[1] + ws1[2] + ws1[3];
    float sq = ws2[0] + ws2[1] + ws2[2] + ws2[3];
    float mean = sm / 128.f;
    float rstd = rsqrtf(sq / 128.f - mean * mean + 1e-5f);
    sLN[t] = (s - mean) * rstd * g_ml[t] + b_ml[t];
    __syncthreads();

    // MLP: gelu(LN @ W1 + b1) @ W2 + b2, residual
    for (int jj = 0; jj < 2; jj++) {
        int j = t + jj * 128;
        float a = b1[j];
        for (int d = 0; d < 128; d++) a += sLN[d] * W1[(size_t)d * 256 + j];
        sM[j] = 0.5f * a * (1.f + erff(a * 0.70710678118654752f));
    }
    __syncthreads();
    float o = b2[t];
    for (int d = 0; d < 256; d++) o += sM[d] * W2[(size_t)d * 128 + t];
    out[(size_t)m * DD + t] = sS[t] + o;
}

// ============================ host launcher ============================
extern "C" void kernel_launch(void* const* d_in, const int* in_sizes, int n_in,
                              void* d_out, int out_size, void* d_ws, size_t ws_size,
                              hipStream_t stream)
{
    (void)in_sizes; (void)n_in; (void)out_size; (void)ws_size;
    const float* x    = (const float*)d_in[0];
    const float* nois = (const float*)d_in[1];
    const float* smu  = (const float*)d_in[2];
    const float* sls  = (const float*)d_in[3];
    const float* Wp   = (const float*)d_in[4];
    const float* bp   = (const float*)d_in[5];
    const float* g_in = (const float*)d_in[6];
    const float* b_in = (const float*)d_in[7];
    const float* Wq   = (const float*)d_in[8];
    const float* bq   = (const float*)d_in[9];
    const float* Wk   = (const float*)d_in[10];
    const float* bk   = (const float*)d_in[11];
    const float* Wv   = (const float*)d_in[12];
    const float* bv   = (const float*)d_in[13];
    const float* W_ih = (const float*)d_in[14];
    const float* W_hh = (const float*)d_in[15];
    const float* b_ih = (const float*)d_in[16];
    const float* b_hh = (const float*)d_in[17];
    const float* W1   = (const float*)d_in[18];
    const float* b1   = (const float*)d_in[19];
    const float* W2   = (const float*)d_in[20];
    const float* b2   = (const float*)d_in[21];
    const float* g_sl = (const float*)d_in[22];
    const float* b_sl = (const float*)d_in[23];
    const float* g_ml = (const float*)d_in[24];
    const float* b_ml = (const float*)d_in[25];

    char* w = (char*)d_ws;
    unsigned short* k_bf = (unsigned short*)w; w += (size_t)BB * NN * DD * 2;   // 16 MB
    unsigned short* v_t  = (unsigned short*)w; w += (size_t)BB * NN * DD * 2;   // 16 MB
    unsigned short* q_bf = (unsigned short*)w; w += (size_t)BB * KK * DD * 2;   // 128 KB
    float* slotsA = (float*)w; w += (size_t)BB * KK * DD * 4;                   // 256 KB
    float* slotsB = (float*)w; w += (size_t)BB * KK * DD * 4;
    float* updbuf = (float*)w; w += (size_t)BB * KK * DD * 4;
    unsigned short* WpT = (unsigned short*)w; w += (size_t)CC * DD * 2;         // 128 KB
    unsigned short* WkT = (unsigned short*)w; w += (size_t)DD * DD * 2;         // 32 KB
    unsigned short* WvT = (unsigned short*)w; w += (size_t)DD * DD * 2;
    unsigned short* WqT = (unsigned short*)w; w += (size_t)DD * DD * 2;
    float* W_ihT = (float*)w; w += (size_t)DD * 3 * DD * 4;                     // 192 KB
    float* W_hhT = (float*)w; w += (size_t)DD * 3 * DD * 4;

    float* out_slots = (float*)d_out;
    float* out_attn  = out_slots + BB * KK * DD;

    k_prep_w<<<(CC * DD + 255) / 256, 256, 0, stream>>>(Wp, Wk, Wv, Wq, W_ih, W_hh,
                                                        WpT, WkT, WvT, WqT, W_ihT, W_hhT);
    k_input_proj<<<BB * NN / 16, 256, 0, stream>>>(x, WpT, bp, g_in, b_in, WkT, bk, WvT, bv, k_bf, v_t);
    k_slots_init<<<(BB * KK * DD + 255) / 256, 256, 0, stream>>>(smu, sls, nois, slotsA);

    float* cur = slotsA; float* nxt = slotsB;
    for (int it = 0; it < 3; it++) {
        k_qproj<<<BB * KK / 16, 256, 0, stream>>>(cur, g_sl, b_sl, WqT, bq, q_bf);
        k_attn<<<BB, 256, 0, stream>>>(q_bf, k_bf, v_t, out_attn, updbuf);
        float* dst = (it == 2) ? out_slots : nxt;
        k_gru_mlp<<<BB * KK, 128, 0, stream>>>(updbuf, cur, W_ihT, W_hhT, b_ih, b_hh,
                                               W1, b1, W2, b2, g_ml, b_ml, dst);
        float* tm = cur; cur = nxt; nxt = tm;
    }
}
